// ConvLSTM_1683627180563
// MI455X (gfx1250) — matrix operand order repaired and checked by reading, not compiled, not run on hardware
//
#include <hip/hip_runtime.h>

typedef __attribute__((ext_vector_type(16))) _Float16 v16h;
typedef __attribute__((ext_vector_type(8)))  float    v8f;

union Frag16 {
  v16h v;
  struct { int4 lo; int4 hi; } q;
};

// ---------------------------------------------------------------- utilities
__global__ __launch_bounds__(256) void zero_f32v4(float* p, size_t n4) {
  size_t i = (size_t)blockIdx.x * 256 + threadIdx.x;
  if (i < n4) ((int4*)p)[i] = make_int4(0, 0, 0, 0);
}

// w: [256][Cc][3][3] fp32  ->  wp: [256][9][CCP] f16 (zero-padded channels)
__global__ __launch_bounds__(256) void pack_weights(const float* __restrict__ w,
                                                    _Float16* __restrict__ wp,
                                                    int Cc, int CCP) {
  int idx = blockIdx.x * 256 + threadIdx.x;
  int total = 256 * 9 * CCP;
  if (idx >= total) return;
  int o   = idx / (9 * CCP);
  int r   = idx - o * 9 * CCP;
  int pos = r / CCP;
  int cp  = r - pos * CCP;
  float v = 0.0f;
  if (cp < Cc) v = w[(size_t)(o * Cc + cp) * 9 + pos];
  wp[idx] = (_Float16)v;
}

// ------------------------------------------------- fused ConvLSTM time step
// One block handles one (image n, row y): computes all 256 gate channels for
// the 64 pixels of that row via WMMA implicit GEMM, then the LSTM pointwise.
template<int CCP, int CX>
__global__ __launch_bounds__(256) void convlstm_step(
    const float* __restrict__ in, size_t in_off, size_t in_nstride,
    const float* __restrict__ hprev,
    float* cstate,                       // in-place c update (elementwise safe)
    const _Float16* __restrict__ wp,
    const float* __restrict__ bias,
    float* __restrict__ hout,
    float* __restrict__ oslot, size_t o_nstride)
{
  constexpr int H = 64, W = 64, HD = 64;
  constexpr int CTOT  = CX + HD;          // real channels (x ++ h)
  constexpr int PITCH = CCP + 8;          // halfs; multiple of 8 -> 16B aligned cols
  constexpr int SLAB_H  = 3 * 66 * PITCH; // 3 rows x 66 padded cols x channels
  constexpr int GP      = 72;             // gate row pitch (halfs)
  constexpr int GATES_H = 256 * GP;
  constexpr int SMEM_H  = SLAB_H > GATES_H ? SLAB_H : GATES_H;
  __shared__ __align__(16) _Float16 smem[SMEM_H];

  const int tid   = threadIdx.x;
  const int wave  = tid >> 5;             // 8 waves; wave owns gate rows [32w,32w+32)
  const int lane  = tid & 31;
  const int l15   = lane & 15;
  const int lhalf = lane >> 4;
  const int n = blockIdx.x >> 6;
  const int y = blockIdx.x & 63;

  // ---- zero slab (covers x/row/channel zero-padding) ----
  for (int i = tid; i < SLAB_H / 8; i += 256)
    ((int4*)smem)[i] = make_int4(0, 0, 0, 0);
  __syncthreads();

  // ---- fill slab: slab[ky][xx][c] = f16(concat(x_t, h_prev)) ----
  {
    const int xcol = tid & 63;
    for (int pc = tid >> 6; pc < 3 * CTOT; pc += 4) {
      const int ky = pc / CTOT;
      const int c  = pc - ky * CTOT;
      const int gy = y + ky - 1;                      // uniform per (pc)
      if (gy >= 0 && gy < H) {
        float v;
        if (c < CX)
          v = in[in_off + (size_t)n * in_nstride + (size_t)c * (H * W) + gy * W + xcol];
        else
          v = hprev[((size_t)(n * HD + (c - CX))) * (H * W) + gy * W + xcol];
        smem[(ky * 66 + xcol + 1) * PITCH + c] = (_Float16)v;
      }
    }
  }
  __syncthreads();

  // ---- implicit-GEMM: gates[256 x 64] = W f16 x patches f16, fp32 accum ----
  v8f acc[2][4] = {};
  for (int pos = 0; pos < 9; ++pos) {
    const int kyb = (pos / 3) * 66;
    const int kx  = pos % 3;
    for (int kc = 0; kc < CCP / 32; ++kc) {
      const int kb = kc * 32;
      Frag16 a[2];
#pragma unroll
      for (int mt = 0; mt < 2; ++mt) {
        // A 16x32 f16 layout: lane-half h holds K = {8h..8h+7} U {16+8h..16+8h+7}
        const _Float16* ap =
            wp + ((size_t)((wave * 32 + mt * 16 + l15) * 9 + pos)) * CCP + kb + lhalf * 8;
        a[mt].q.lo = *(const int4*)ap;
        a[mt].q.hi = *(const int4*)(ap + 16);
      }
      Frag16 b[4];
#pragma unroll
      for (int nt = 0; nt < 4; ++nt) {
        // B 32x16 f16 layout: lane-half h holds K = {8h..8h+7} U {16+8h..16+8h+7} for col = lane&15
        const _Float16* bp =
            &smem[(kyb + nt * 16 + l15 + kx) * PITCH + kb + lhalf * 8];
        b[nt].q.lo = *(const int4*)bp;
        b[nt].q.hi = *(const int4*)(bp + 16);
      }
#pragma unroll
      for (int mt = 0; mt < 2; ++mt)
#pragma unroll
        for (int nt = 0; nt < 4; ++nt)
          acc[mt][nt] = __builtin_amdgcn_wmma_f32_16x16x32_f16(
              false, a[mt].v, false, b[nt].v, (short)0, acc[mt][nt], false, false);
    }
  }
  __syncthreads();  // slab dead; reuse smem for gates

  // ---- spill gates (f16) to LDS. C layout: VGPR r -> M = r + 8*lanehalf ----
#pragma unroll
  for (int mt = 0; mt < 2; ++mt)
#pragma unroll
    for (int nt = 0; nt < 4; ++nt) {
      const int row0 = wave * 32 + mt * 16 + lhalf * 8;
      const int col  = nt * 16 + l15;
#pragma unroll
      for (int r = 0; r < 8; ++r)
        smem[(row0 + r) * GP + col] = (_Float16)acc[mt][nt][r];
    }
  __syncthreads();

  // ---- LSTM pointwise: i,f,o,g gate split along channel blocks of 64 ----
  for (int idx = tid; idx < HD * W; idx += 256) {
    const int hd = idx >> 6;
    const int p  = idx & 63;
    const float gi = (float)smem[(hd      ) * GP + p] + bias[hd      ];
    const float gf = (float)smem[(hd +  64) * GP + p] + bias[hd +  64];
    const float go = (float)smem[(hd + 128) * GP + p] + bias[hd + 128];
    const float gg = (float)smem[(hd + 192) * GP + p] + bias[hd + 192];
    const float si = 1.0f / (1.0f + expf(-gi));
    const float sf = 1.0f / (1.0f + expf(-gf));
    const float so = 1.0f / (1.0f + expf(-go));
    const float tg = tanhf(gg);
    const size_t off = ((size_t)(n * HD + hd)) * (H * W) + y * W + p;
    const float cold = cstate[off];
    const float cn = sf * cold + si * tg;
    const float hn = so * tanhf(cn);
    cstate[off] = cn;
    hout[off]   = hn;
    if (oslot)
      oslot[(size_t)n * o_nstride + (size_t)hd * (H * W) + y * W + p] = hn;
  }
}

// ---------------------------------------------------------------- launcher
extern "C" void kernel_launch(void* const* d_in, const int* in_sizes, int n_in,
                              void* d_out, int out_size, void* d_ws, size_t ws_size,
                              hipStream_t stream) {
  const float* x  = (const float*)d_in[0];
  const float* w0 = (const float*)d_in[1];
  const float* b0 = (const float*)d_in[2];
  const float* w1 = (const float*)d_in[3];
  const float* b1 = (const float*)d_in[4];
  float* out = (float*)d_out;

  constexpr size_t HW   = 64 * 64;
  constexpr size_t CHW  = 64 * HW;          // one [Hd,H,W] slab = 262144
  constexpr size_t OUT1 = 8ull * 16 * CHW;  // out1 stack

  float* out1 = out;
  float* h0o  = out + OUT1;        // final h0
  float* c0   = h0o + 8 * CHW;     // c0 state, updated in place
  float* h1o  = c0  + 8 * CHW;     // final h1
  float* c1   = h1o + 8 * CHW;     // c1 state, updated in place

  char* ws = (char*)d_ws;
  _Float16* wp0 = (_Float16*)ws;                                 // 256*9*96  f16
  _Float16* wp1 = (_Float16*)(ws + (size_t)256 * 9 * 96 * 2);    // 256*9*128 f16
  float* h0w = (float*)(ws + (size_t)256 * 9 * 96 * 2 + (size_t)256 * 9 * 128 * 2);
  float* h1w = (float*)((char*)h0w + 8 * CHW * 4);

  // zero the 4 state regions (h0,c0,h1,c1 = 4 * 8*CHW floats = 2M int4)
  zero_f32v4<<<(2097152 + 255) / 256, 256, 0, stream>>>(h0o, 2097152);
  pack_weights<<<(256 * 9 * 96  + 255) / 256, 256, 0, stream>>>(w0, wp0, 80, 96);
  pack_weights<<<(256 * 9 * 128 + 255) / 256, 256, 0, stream>>>(w1, wp1, 128, 128);

  for (int t = 0; t < 16; ++t) {
    // h ping-pong: odd steps write the d_out region -> t=15 leaves final h there
    float* h0_in = (t & 1) ? h0w : h0o;
    float* h0_ot = (t & 1) ? h0o : h0w;
    convlstm_step<96, 16><<<512, 256, 0, stream>>>(
        x, (size_t)t * 16 * HW, (size_t)16 * 16 * HW,
        h0_in, c0, wp0, b0, h0_ot, (float*)nullptr, 0);

    float* h1_in = (t & 1) ? h1w : h1o;
    float* h1_ot = (t & 1) ? h1o : h1w;
    convlstm_step<128, 64><<<512, 256, 0, stream>>>(
        h0_ot, 0, CHW,
        h1_in, c1, wp1, b1, h1_ot,
        out1 + (size_t)t * CHW, (size_t)16 * CHW);
  }
}
